// ContrastiveLoss_86912958202422
// MI455X (gfx1250) — compile-verified, matrix operand order checked by reading, not run on hardware
//
#include <hip/hip_runtime.h>
#include <hip/hip_bf16.h>
#include <math.h>

// ---------------------------------------------------------------------------
// ContrastiveLoss on MI455X (gfx1250): fused bf16-WMMA GEMM + masked reduction
// Software-pipelined K-loop: next-step A/B fragments prefetched while current
// step's v_wmma_f32_16x16x32_bf16 instructions execute.
// ---------------------------------------------------------------------------

typedef __attribute__((ext_vector_type(16))) __bf16 bf16x16;
typedef __attribute__((ext_vector_type(8)))  float  v8f;

union FragU {
    bf16x16 v;
    uint4   q[2];
};

#define D_DIM   256
#define TILE_M  128
#define TILE_N  64
#define NBINS   4096

__device__ __forceinline__ unsigned short f32_to_bf16(float f) {
    unsigned int u = __float_as_uint(f);
    u += 0x7FFFu + ((u >> 16) & 1u);   // round-to-nearest-even
    return (unsigned short)(u >> 16);
}

// ---------------------------------------------------------------------------
// Kernel 0: zero accumulators + histogram region
// ---------------------------------------------------------------------------
__global__ void zero_kernel(int* __restrict__ p, int n) {
    int i = blockIdx.x * blockDim.x + threadIdx.x;
    if (i < n) p[i] = 0;
}

// ---------------------------------------------------------------------------
// Kernel 1: L2-normalize rows of the 3 embeddings, emit bf16 z[3][N][256]
// One wave32 per row; 8 floats per lane.
// ---------------------------------------------------------------------------
__global__ __launch_bounds__(256) void normalize_kernel(
    const float* __restrict__ e0, const float* __restrict__ e1,
    const float* __restrict__ e2, unsigned short* __restrict__ zout, int N)
{
    const int gw   = (blockIdx.x * blockDim.x + threadIdx.x) >> 5;
    const int lane = threadIdx.x & 31;
    if (gw >= 3 * N) return;
    const int which = gw / N;
    const int row   = gw - which * N;
    const float* src = (which == 0) ? e0 : (which == 1) ? e1 : e2;
    src += (long long)row * D_DIM;

    float4 v0 = ((const float4*)src)[lane * 2 + 0];
    float4 v1 = ((const float4*)src)[lane * 2 + 1];

    float ss = v0.x * v0.x + v0.y * v0.y + v0.z * v0.z + v0.w * v0.w +
               v1.x * v1.x + v1.y * v1.y + v1.z * v1.z + v1.w * v1.w;
    #pragma unroll
    for (int off = 16; off > 0; off >>= 1) ss += __shfl_xor(ss, off, 32);

    const float scale = 1.0f / fmaxf(sqrtf(ss), 1e-12f);

    unsigned short o[8];
    o[0] = f32_to_bf16(v0.x * scale); o[1] = f32_to_bf16(v0.y * scale);
    o[2] = f32_to_bf16(v0.z * scale); o[3] = f32_to_bf16(v0.w * scale);
    o[4] = f32_to_bf16(v1.x * scale); o[5] = f32_to_bf16(v1.y * scale);
    o[6] = f32_to_bf16(v1.z * scale); o[7] = f32_to_bf16(v1.w * scale);

    unsigned short* dst = zout + ((long long)which * N + row) * D_DIM + lane * 8;
    *(uint4*)dst = *(const uint4*)o;   // 16B store, 16B aligned
}

// ---------------------------------------------------------------------------
// Kernel 2: histogram of batch indices (values in [0, 4096))
// ---------------------------------------------------------------------------
__global__ void hist_kernel(const int* __restrict__ batch, int* __restrict__ hist, int N) {
    int i = blockIdx.x * blockDim.x + threadIdx.x;
    if (i < N) atomicAdd(&hist[batch[i] & (NBINS - 1)], 1);
}

// Kernel 3: pos_cnt = sum(c^2) - N  ->  accum[6]
__global__ __launch_bounds__(256) void hist_reduce_kernel(
    const int* __restrict__ hist, float* __restrict__ accum, int N)
{
    __shared__ float red[256];
    float s = 0.0f;
    for (int b = threadIdx.x; b < NBINS; b += 256) {
        float c = (float)hist[b];
        s += c * c;
    }
    red[threadIdx.x] = s;
    __syncthreads();
    for (int off = 128; off > 0; off >>= 1) {
        if ((int)threadIdx.x < off) red[threadIdx.x] += red[threadIdx.x + off];
        __syncthreads();
    }
    if (threadIdx.x == 0) accum[6] = red[0] - (float)N;
}

// ---------------------------------------------------------------------------
// Kernel 4: fused sim-GEMM + masked loss reduction.
// Block: 256 threads (8 waves). Tile: 128(M) x 64(N). blockIdx.z = pair id.
// Each wave: 16 rows x 64 cols = 4x v_wmma_f32_16x16x32_bf16 accumulators,
// K-loop over D=256 in steps of 32, software-pipelined (prefetch next step).
// ---------------------------------------------------------------------------
__global__ __launch_bounds__(256) void gemm_loss_kernel(
    const unsigned short* __restrict__ z,   // bf16 bits, [3][N][256]
    const int* __restrict__ batch,
    float* __restrict__ partials,           // [3 * (N/128) * (N/64)][2]
    int N)
{
    __shared__ unsigned short Bs[TILE_N * D_DIM];   // 32 KB
    __shared__ int   biRow[TILE_M];
    __shared__ int   bjCol[TILE_N];
    __shared__ float red[16];

    const int tid  = threadIdx.x;
    const int lane = tid & 31;
    const int wave = tid >> 5;                      // 0..7
    const int p    = blockIdx.z;                    // 0:(0,1) 1:(0,2) 2:(1,2)
    const int ia   = (p == 2) ? 1 : 0;
    const int ib   = (p == 0) ? 1 : 2;
    const long long NS = (long long)N * D_DIM;
    const unsigned short* zi = z + (long long)ia * NS;
    const unsigned short* zj = z + (long long)ib * NS;

    const int rowBase = blockIdx.y * TILE_M;
    const int colBase = blockIdx.x * TILE_N;

    // Stage B tile (64 x 256 bf16) into LDS, 16B chunks
    for (int c = tid; c < TILE_N * D_DIM / 8; c += 256) {
        const int e = c * 8;
        const int r = e >> 8;          // /256
        const int k = e & 255;
        *(uint4*)(&Bs[e]) =
            *(const uint4*)(zj + (long long)(colBase + r) * D_DIM + k);
    }
    if (tid < TILE_M) biRow[tid] = batch[rowBase + tid];
    if (tid < TILE_N) bjCol[tid] = batch[colBase + tid];
    __syncthreads();

    const int half = lane >> 4;
    const int l16  = lane & 15;

    // Fragment base pointers (16-bit A 16x32 layout: lane half picks K range)
    const unsigned short* arow =
        zi + (long long)(rowBase + wave * 16 + l16) * D_DIM + 8 * half;
    const unsigned short* brow = &Bs[l16 * D_DIM + 8 * half];

    v8f acc[4];
    acc[0] = (v8f)(0.0f); acc[1] = (v8f)(0.0f);
    acc[2] = (v8f)(0.0f); acc[3] = (v8f)(0.0f);

    // ---- software pipeline: prologue loads K-step 0 ----
    FragU a_cur, b_cur[4];
    a_cur.q[0] = *(const uint4*)(arow + 0);
    a_cur.q[1] = *(const uint4*)(arow + 16);
    #pragma unroll
    for (int nt = 0; nt < 4; ++nt) {
        const unsigned short* bp = brow + nt * 16 * D_DIM;
        b_cur[nt].q[0] = *(const uint4*)(bp + 0);
        b_cur[nt].q[1] = *(const uint4*)(bp + 16);
    }

    #pragma unroll
    for (int k0 = 0; k0 < D_DIM; k0 += 32) {
        const int kn = k0 + 32;
        FragU a_nxt, b_nxt[4];
        if (kn < D_DIM) {
            // Prefetch next K-step fragments (issued before current WMMAs)
            a_nxt.q[0] = *(const uint4*)(arow + kn);
            a_nxt.q[1] = *(const uint4*)(arow + kn + 16);
            #pragma unroll
            for (int nt = 0; nt < 4; ++nt) {
                const unsigned short* bp = brow + nt * 16 * D_DIM + kn;
                b_nxt[nt].q[0] = *(const uint4*)(bp + 0);
                b_nxt[nt].q[1] = *(const uint4*)(bp + 16);
            }
        }

        #pragma unroll
        for (int nt = 0; nt < 4; ++nt) {
            acc[nt] = __builtin_amdgcn_wmma_f32_16x16x32_bf16(
                false, a_cur.v, false, b_cur[nt].v, (short)0, acc[nt],
                false, false);
        }

        if (kn < D_DIM) {
            a_cur = a_nxt;
            #pragma unroll
            for (int nt = 0; nt < 4; ++nt) b_cur[nt] = b_nxt[nt];
        }
    }

    // Epilogue: sim = dot / T; masked accumulation.
    // C/D layout: lane l, VGPR r -> M = 8*(l>>4)+r, N = l&15.
    float posS = 0.0f, negS = 0.0f;
    #pragma unroll
    for (int nt = 0; nt < 4; ++nt) {
        #pragma unroll
        for (int r = 0; r < 8; ++r) {
            const int mloc = wave * 16 + 8 * half + r;
            const int nloc = nt * 16 + l16;
            const float sim = acc[nt][r] * 2.0f;            // 1/TEMPERATURE
            const bool pos = (biRow[mloc] == bjCol[nloc]) &&
                             ((rowBase + mloc) != (colBase + nloc));
            if (pos) posS += sim;
            else     negS += fmaxf(1.0f + sim, 0.0f);       // MARGIN = 1
        }
    }

    // wave32 shuffle reduce, then block reduce -> deterministic partials
    #pragma unroll
    for (int off = 16; off > 0; off >>= 1) {
        posS += __shfl_xor(posS, off, 32);
        negS += __shfl_xor(negS, off, 32);
    }
    if (lane == 0) { red[wave] = posS; red[8 + wave] = negS; }
    __syncthreads();
    if (tid == 0) {
        float P = 0.0f, Ng = 0.0f;
        #pragma unroll
        for (int w = 0; w < 8; ++w) { P += red[w]; Ng += red[8 + w]; }
        const long long bid =
            ((long long)blockIdx.z * gridDim.y + blockIdx.y) * gridDim.x + blockIdx.x;
        partials[2 * bid + 0] = P;
        partials[2 * bid + 1] = Ng;
    }
}

// ---------------------------------------------------------------------------
// Kernel 5: deterministic final reduction over per-block partials
// ---------------------------------------------------------------------------
__global__ __launch_bounds__(256) void finalize_kernel(
    const float* __restrict__ partials, const float* __restrict__ accum,
    float* __restrict__ out, int N, int blocksPerPair)
{
    __shared__ float redP[256];
    __shared__ float redN[256];
    const float pos_cnt = accum[6];
    const float neg_cnt = (float)N * (float)N - pos_cnt;   // neg mask includes diagonal
    float loss = 0.0f;
    for (int p = 0; p < 3; ++p) {
        float sp = 0.0f, sn = 0.0f;
        for (int b = threadIdx.x; b < blocksPerPair; b += 256) {
            const long long bid = (long long)p * blocksPerPair + b;
            sp += partials[2 * bid + 0];
            sn += partials[2 * bid + 1];
        }
        redP[threadIdx.x] = sp;
        redN[threadIdx.x] = sn;
        __syncthreads();
        for (int off = 128; off > 0; off >>= 1) {
            if ((int)threadIdx.x < off) {
                redP[threadIdx.x] += redP[threadIdx.x + off];
                redN[threadIdx.x] += redN[threadIdx.x + off];
            }
            __syncthreads();
        }
        if (threadIdx.x == 0)
            loss += (-redP[0] / pos_cnt + redN[0] / neg_cnt);
        __syncthreads();
    }
    if (threadIdx.x == 0) out[0] = loss * (1.0f / 3.0f);
}

// ---------------------------------------------------------------------------
// Host launcher
// ---------------------------------------------------------------------------
extern "C" void kernel_launch(void* const* d_in, const int* in_sizes, int n_in,
                              void* d_out, int out_size, void* d_ws, size_t ws_size,
                              hipStream_t stream) {
    (void)n_in; (void)out_size; (void)ws_size;

    const float* e0    = (const float*)d_in[0];
    const float* e1    = (const float*)d_in[1];
    const float* e2    = (const float*)d_in[2];
    const int*   batch = (const int*)d_in[3];
    const int N = in_sizes[3];                       // 8192

    // Workspace layout (bytes):
    //   [0,      64)     float accum[16]      (accum[6] = pos_cnt)
    //   [64,     16448)  int   hist[4096]
    //   [16640,  213248) float partials[2 * 3*(N/128)*(N/64)]
    //   [262144, ...)    bf16  z[3][N][256]   (~12 MB)
    char* ws = (char*)d_ws;
    float*          accum    = (float*)ws;
    float*          partials = (float*)(ws + 16640);
    unsigned short* zbuf     = (unsigned short*)(ws + 262144);

    // 0) zero accum + histogram (16448 bytes = 4112 ints, contiguous from ws)
    zero_kernel<<<(4112 + 255) / 256, 256, 0, stream>>>((int*)ws, 4112);

    // 1) normalize -> bf16 z
    {
        const int waves   = 3 * N;
        const int threads = waves * 32;
        normalize_kernel<<<(threads + 255) / 256, 256, 0, stream>>>(e0, e1, e2, zbuf, N);
    }

    // 2) histogram + pos_cnt
    hist_kernel<<<(N + 255) / 256, 256, 0, stream>>>(batch, (int*)(ws + 64), N);
    hist_reduce_kernel<<<1, 256, 0, stream>>>((const int*)(ws + 64), accum, N);

    // 3) fused WMMA GEMM + masked loss partials
    {
        dim3 grid(N / TILE_N, N / TILE_M, 3);        // (128, 64, 3)
        gemm_loss_kernel<<<grid, 256, 0, stream>>>(zbuf, batch, partials, N);
    }

    // 4) final deterministic reduction -> scalar loss
    {
        const int blocksPerPair = (N / TILE_N) * (N / TILE_M);
        finalize_kernel<<<1, 256, 0, stream>>>(partials, accum, (float*)d_out, N, blocksPerPair);
    }
}
